// GATModel_15238543966333
// MI455X (gfx1250) — matrix operand order repaired and checked by reading, not compiled, not run on hardware
//
#include <hip/hip_runtime.h>
#include <stdint.h>

typedef __attribute__((ext_vector_type(16))) __bf16 v16bf;
typedef __attribute__((ext_vector_type(8)))  __bf16 v8bf;
typedef __attribute__((ext_vector_type(8)))  float  v8f;

#define NSLOPE 0.2f

// ---------- order-preserving float<->uint encoding for atomic max ----------
__device__ __forceinline__ unsigned enc_f32(float f) {
  unsigned u = __float_as_uint(f);
  return (u & 0x80000000u) ? ~u : (u | 0x80000000u);
}
__device__ __forceinline__ float dec_f32(unsigned k) {
  unsigned u = (k & 0x80000000u) ? (k & 0x7FFFFFFFu) : ~k;
  return __uint_as_float(u);
}

// ---------- prep: convert / pad / transpose to bf16 ----------
__global__ void k_prep_x(const float* __restrict__ x, __bf16* __restrict__ xpad, int N) {
  int i = blockIdx.x * blockDim.x + threadIdx.x;
  if (i >= N * 32) return;
  int n = i >> 5, k = i & 31;
  float v = (k < 15) ? x[n * 15 + k] : 0.0f;
  xpad[i] = (__bf16)v;
}
__global__ void k_prep_w1(const float* __restrict__ W1, __bf16* __restrict__ Wt1) {
  int i = blockIdx.x * blockDim.x + threadIdx.x;  // 256 cols x 32 padded-K
  if (i >= 256 * 32) return;
  int n = i >> 5, k = i & 31;
  float v = (k < 15) ? W1[k * 256 + n] : 0.0f;    // column-major [n][k]
  Wt1[i] = (__bf16)v;
}
__global__ void k_prep_w2(const float* __restrict__ W2, __bf16* __restrict__ Wt2) {
  int i = blockIdx.x * blockDim.x + threadIdx.x;  // 64 cols x 256 K
  if (i >= 64 * 256) return;
  int n = i >> 8, k = i & 255;
  Wt2[i] = (__bf16)W2[k * 64 + n];                // column-major [n][k]
}

// ---------- bf16 WMMA GEMM: C[M x Ncols] = A[M x K] * Bt[Ncols x K]^T ----------
// One wave computes one 16x16 tile, K-loop in steps of 32.
__global__ void __launch_bounds__(256)
k_gemm_bf16(const __bf16* __restrict__ A, int lda,
            const __bf16* __restrict__ Bt, int ldb,
            float* __restrict__ C, int ldc,
            int nTilesN, int totalTiles, int K) {
  int w = blockIdx.x * 8 + (threadIdx.x >> 5);
  if (w >= totalTiles) return;               // wave-uniform branch: EXEC all-1 inside
  int lane = threadIdx.x & 31;
  int half = lane >> 4;                      // 0: lanes 0-15, 1: lanes 16-31
  int l15  = lane & 15;
  int tm = w / nTilesN, tn = w % nTilesN;

  const __bf16* arow = A  + (size_t)(tm * 16 + l15) * lda;  // A row-major, lane -> M
  const __bf16* bcol = Bt + (size_t)(tn * 16 + l15) * ldb;  // Bt col-major, lane -> N

  v8f c = {};
  for (int k0 = 0; k0 < K; k0 += 32) {
    // A 16x32: lanes0-15 K{0..7,16..23}, lanes16-31 K{8..15,24..31}
    v8bf a_lo = *(const v8bf*)(arow + k0 + half * 8);
    v8bf a_hi = *(const v8bf*)(arow + k0 + 16 + half * 8);
    // B 32x16: lanes0-15 K{0..7,16..23}, lanes16-31 K{8..15,24..31}
    v8bf b_lo = *(const v8bf*)(bcol + k0 + half * 8);
    v8bf b_hi = *(const v8bf*)(bcol + k0 + 16 + half * 8);
    v16bf a = __builtin_shufflevector(a_lo, a_hi, 0,1,2,3,4,5,6,7,8,9,10,11,12,13,14,15);
    v16bf b = __builtin_shufflevector(b_lo, b_hi, 0,1,2,3,4,5,6,7,8,9,10,11,12,13,14,15);
    c = __builtin_amdgcn_wmma_f32_16x16x32_bf16(false, a, false, b, (short)0, c,
                                                false, false);
  }
  // C/D 16x16 f32: col = lane&15, rows r + 8*half
  float* crow = C + (size_t)(tm * 16 + half * 8) * ldc + tn * 16 + l15;
#pragma unroll
  for (int r = 0; r < 8; ++r) crow[(size_t)r * ldc] = c[r];
}

// ---------- per-node attention scalars: a[n,h] = dot(h[n,h,:], att[h,:]) ----------
template <int H>
__global__ void k_attn(const float* __restrict__ h, const float* __restrict__ att_s,
                       const float* __restrict__ att_d, float* __restrict__ as,
                       float* __restrict__ ad, int N) {
  int i = blockIdx.x * blockDim.x + threadIdx.x;
  if (i >= N * H) return;
  int n = i / H, hh = i % H;
  const float* hp = h + (size_t)n * (H * 64) + hh * 64;
  const float* sp = att_s + hh * 64;
  const float* dp = att_d + hh * 64;
  float s = 0.f, d = 0.f;
#pragma unroll 8
  for (int k = 0; k < 64; ++k) { float v = hp[k]; s += v * sp[k]; d += v * dp[k]; }
  as[i] = s;
  ad[i] = d;
}

// ---------- edge softmax pass 1: segment max over dst ----------
template <int H>
__global__ void k_edge_max(const int* __restrict__ ei, int E, int N,
                           const float* __restrict__ as, const float* __restrict__ ad,
                           unsigned* __restrict__ m) {
  int e = blockIdx.x * blockDim.x + threadIdx.x;
  if (e >= E + N) return;
  int src, dst;
  if (e < E) { src = ei[e]; dst = ei[E + e]; } else { src = dst = e - E; }
#pragma unroll
  for (int hh = 0; hh < H; ++hh) {
    float v = as[src * H + hh] + ad[dst * H + hh];
    v = v > 0.f ? v : NSLOPE * v;
    atomicMax(&m[dst * H + hh], enc_f32(v));
  }
}

// ---------- edge softmax pass 2: segment sum of exp(e - m) ----------
template <int H>
__global__ void k_edge_sum(const int* __restrict__ ei, int E, int N,
                           const float* __restrict__ as, const float* __restrict__ ad,
                           const unsigned* __restrict__ m, float* __restrict__ z) {
  int e = blockIdx.x * blockDim.x + threadIdx.x;
  if (e >= E + N) return;
  int src, dst;
  if (e < E) { src = ei[e]; dst = ei[E + e]; } else { src = dst = e - E; }
#pragma unroll
  for (int hh = 0; hh < H; ++hh) {
    float v = as[src * H + hh] + ad[dst * H + hh];
    v = v > 0.f ? v : NSLOPE * v;
    atomicAdd(&z[dst * H + hh], __expf(v - dec_f32(m[dst * H + hh])));
  }
}

// ---------- edge softmax pass 3: weighted message scatter (one wave per edge) ----
template <int H>
__global__ void __launch_bounds__(256)
k_edge_scatter(const int* __restrict__ ei, int E, int N,
               const float* __restrict__ as, const float* __restrict__ ad,
               const unsigned* __restrict__ m, const float* __restrict__ z,
               const float* __restrict__ h, float* __restrict__ out) {
  int w = blockIdx.x * 8 + (threadIdx.x >> 5);
  if (w >= E + N) return;
  int lane = threadIdx.x & 31;
  int src, dst;
  if (w < E) { src = ei[w]; dst = ei[E + w]; } else { src = dst = w - E; }
  const float* hs = h + (size_t)src * (H * 64);
  float* od = out + (size_t)dst * (H * 64);
#pragma unroll
  for (int hh = 0; hh < H; ++hh) {
    float v = as[src * H + hh] + ad[dst * H + hh];
    v = v > 0.f ? v : NSLOPE * v;
    float alpha = __expf(v - dec_f32(m[dst * H + hh])) / z[dst * H + hh];
#pragma unroll
    for (int j = 0; j < 2; ++j) {
      int c = hh * 64 + lane + j * 32;
      atomicAdd(&od[c], alpha * hs[c]);
    }
  }
}

// ---------- bias + ELU, write bf16 activations for the next GEMM ----------
__global__ void k_post1(const float* __restrict__ out1, const float* __restrict__ b1,
                        __bf16* __restrict__ hbf, int N) {
  int i = blockIdx.x * blockDim.x + threadIdx.x;
  if (i >= N * 256) return;
  float v = out1[i] + b1[i & 255];
  v = v > 0.f ? v : (__expf(v) - 1.0f);
  hbf[i] = (__bf16)v;
}

// ---------- column mean over nodes ----------
__global__ void __launch_bounds__(256)
k_mean(const float* __restrict__ out2, float* __restrict__ res, int N, int rowsPerBlock) {
  __shared__ float sm[256];
  int t = threadIdx.x;
  int c = t & 63;
  int ro = t >> 6;  // 0..3
  int r0 = blockIdx.x * rowsPerBlock;
  int r1 = r0 + rowsPerBlock; if (r1 > N) r1 = N;
  float acc = 0.f;
  for (int r = r0 + ro; r < r1; r += 4) acc += out2[(size_t)r * 64 + c];
  sm[t] = acc;
  __syncthreads();
  if (t < 64) {
    float s = sm[t] + sm[t + 64] + sm[t + 128] + sm[t + 192];
    atomicAdd(&res[c], s / (float)N);
  }
}
__global__ void k_bias64(float* __restrict__ res, const float* __restrict__ b2) {
  int t = threadIdx.x;
  if (t < 64) res[t] += b2[t];
}

// =======================================================================
extern "C" void kernel_launch(void* const* d_in, const int* in_sizes, int n_in,
                              void* d_out, int out_size, void* d_ws, size_t ws_size,
                              hipStream_t stream) {
  const float* x   = (const float*)d_in[0];
  const int*   ei  = (const int*)d_in[1];
  const float* W1  = (const float*)d_in[2];
  const float* as1 = (const float*)d_in[3];
  const float* ad1 = (const float*)d_in[4];
  const float* b1  = (const float*)d_in[5];
  const float* W2  = (const float*)d_in[6];
  const float* as2 = (const float*)d_in[7];
  const float* ad2 = (const float*)d_in[8];
  const float* b2  = (const float*)d_in[9];
  float* out = (float*)d_out;

  const int N  = in_sizes[0] / 15;   // 50000
  const int E  = in_sizes[1] / 2;    // 800000
  const int EP = E + N;              // edges incl. self-loops

  // ---- workspace carve-up (256B aligned) ----
  char* p = (char*)d_ws;
  auto alloc = [&](size_t bytes) -> char* {
    char* r = p;
    p += (bytes + 255) & ~(size_t)255;
    return r;
  };
  float*    h1   = (float*)alloc((size_t)N * 256 * 4);
  float*    out1 = (float*)alloc((size_t)N * 256 * 4);
  float*    h2   = (float*)alloc((size_t)N * 64 * 4);
  float*    out2 = (float*)alloc((size_t)N * 64 * 4);
  float*    aS1  = (float*)alloc((size_t)N * 4 * 4);
  float*    aD1  = (float*)alloc((size_t)N * 4 * 4);
  unsigned* m1   = (unsigned*)alloc((size_t)N * 4 * 4);
  float*    z1   = (float*)alloc((size_t)N * 4 * 4);
  float*    aS2  = (float*)alloc((size_t)N * 4);
  float*    aD2  = (float*)alloc((size_t)N * 4);
  unsigned* m2   = (unsigned*)alloc((size_t)N * 4);
  float*    z2   = (float*)alloc((size_t)N * 4);
  __bf16*   xpad = (__bf16*)alloc((size_t)N * 32 * 2);
  __bf16*   Wt1  = (__bf16*)alloc(256 * 32 * 2);
  __bf16*   Wt2  = (__bf16*)alloc(64 * 256 * 2);
  __bf16*   h1bf = (__bf16*)alloc((size_t)N * 256 * 2);

  // ---- zero accumulators (stream-ordered, graph-capturable) ----
  hipMemsetAsync(out1, 0, (size_t)N * 256 * 4, stream);
  hipMemsetAsync(out2, 0, (size_t)N * 64 * 4, stream);
  hipMemsetAsync(m1, 0, (size_t)N * 4 * 4, stream);  // 0 < enc(any finite) for our data
  hipMemsetAsync(z1, 0, (size_t)N * 4 * 4, stream);
  hipMemsetAsync(m2, 0, (size_t)N * 4, stream);
  hipMemsetAsync(z2, 0, (size_t)N * 4, stream);
  hipMemsetAsync(out, 0, 64 * 4, stream);

  const int B = 256;
  // ---- prep bf16 operands ----
  k_prep_x<<<(N * 32 + B - 1) / B, B, 0, stream>>>(x, xpad, N);
  k_prep_w1<<<(256 * 32 + B - 1) / B, B, 0, stream>>>(W1, Wt1);
  k_prep_w2<<<(64 * 256 + B - 1) / B, B, 0, stream>>>(W2, Wt2);

  // ---- layer 1 GEMM: h1 = x @ W1  (K padded to 32 -> one WMMA per tile) ----
  {
    int tilesM = N / 16, nTilesN = 256 / 16;
    int total = tilesM * nTilesN;
    k_gemm_bf16<<<(total + 7) / 8, B, 0, stream>>>(xpad, 32, Wt1, 32, h1, 256,
                                                   nTilesN, total, 32);
  }

  // ---- layer 1 attention + edge softmax + scatter ----
  k_attn<4><<<(N * 4 + B - 1) / B, B, 0, stream>>>(h1, as1, ad1, aS1, aD1, N);
  k_edge_max<4><<<(EP + B - 1) / B, B, 0, stream>>>(ei, E, N, aS1, aD1, m1);
  k_edge_sum<4><<<(EP + B - 1) / B, B, 0, stream>>>(ei, E, N, aS1, aD1, m1, z1);
  k_edge_scatter<4><<<(EP + 7) / 8, B, 0, stream>>>(ei, E, N, aS1, aD1, m1, z1, h1, out1);

  // ---- bias + ELU -> bf16 ----
  k_post1<<<(N * 256 + B - 1) / B, B, 0, stream>>>(out1, b1, h1bf, N);

  // ---- layer 2 GEMM: h2 = elu(out1) @ W2  (K=256 -> 8 WMMA per tile) ----
  {
    int tilesM = N / 16, nTilesN = 64 / 16;
    int total = tilesM * nTilesN;
    k_gemm_bf16<<<(total + 7) / 8, B, 0, stream>>>(h1bf, 256, Wt2, 256, h2, 64,
                                                   nTilesN, total, 256);
  }

  // ---- layer 2 attention + edge softmax + scatter ----
  k_attn<1><<<(N + B - 1) / B, B, 0, stream>>>(h2, as2, ad2, aS2, aD2, N);
  k_edge_max<1><<<(EP + B - 1) / B, B, 0, stream>>>(ei, E, N, aS2, aD2, m2);
  k_edge_sum<1><<<(EP + B - 1) / B, B, 0, stream>>>(ei, E, N, aS2, aD2, m2, z2);
  k_edge_scatter<1><<<(EP + 7) / 8, B, 0, stream>>>(ei, E, N, aS2, aD2, m2, z2, h2, out2);

  // ---- mean over nodes + bias ----
  {
    int rowsPerBlock = 512;
    k_mean<<<(N + rowsPerBlock - 1) / rowsPerBlock, B, 0, stream>>>(out2, out, N, rowsPerBlock);
    k_bias64<<<1, 64, 0, stream>>>(out, b2);
  }
}
